// TensorODEBLOCK_17935783428407
// MI455X (gfx1250) — matrix operand order repaired and checked by reading, not compiled, not run on hardware
//
#include <hip/hip_runtime.h>
#include <hip/hip_bf16.h>

// ---------------------------------------------------------------------------
// TensorODE block for MI455X (gfx1250, wave32, WMMA f16->f32).
//
//   A0 = x @ P
//   dA/dt = [1,A,A^2,A^3]_flat @ U_flat        (one 2048x2048x512 GEMM/f-eval)
//   RK4 fixed step, NSTEPS=24, t in [0,1]
//   out = A(1) @ F
//
// f16 operands kept pre-swizzled in WMMA fragment layout so every wave loads
// its A/B fragments as contiguous 32B per lane. U scaled by 2^11 and F by 2^9
// (exact) before f16 conversion to dodge f16 subnormals; undone in the f32
// GEMM epilogue.
// ---------------------------------------------------------------------------

typedef __attribute__((ext_vector_type(16))) _Float16 v16h;
typedef __attribute__((ext_vector_type(8)))  float    v8f;

#define B_SZ   2048
#define D_IN   512
#define D_SZ   512
#define D_OUT  512
#define KPOW   4            // poly_dim + 1
#define KBIG   (D_SZ * KPOW) // 2048, basis/U inner dimension
#define NSTEPS 24

// ---------------------------------------------------------------------------
// Fragment-swizzled layouts (per CDNA5 ISA 7.12.2, wave32):
//
// A-operand tile (16 rows m x 32 cols k), 512 f16 = 1KB per tile:
//   lane L (0..31): m = 16*mt + (L&15), g = L>>4
//   stores 16 f16: k = 32*kt + 8*g + j      (j=0..7)   -> VGPR0..3
//                  k = 32*kt + 16 + 8*g + j (j=0..7)   -> VGPR4..7
//   tile order: tileIdx = mt*(K/32) + kt, lane chunk at tileIdx*512 + L*16
//
// B-operand tile (32 rows k x 16 cols n), 512 f16 = 1KB per tile:
//   lane L: n = 16*nt + (L&15), holds k = 32*kt + 8*(L>>4) + j (j=0..7), 32*kt + 16 + 8*(L>>4) + j (j=0..7)
//   tile order: tileIdx = nt*(K/32) + kt
// ---------------------------------------------------------------------------

// fp32 row-major [M,K] -> f16 A-swizzled
__global__ __launch_bounds__(256) void conv_a_swz(const float* __restrict__ src,
                                                  _Float16* __restrict__ dst, int K) {
    int gid  = blockIdx.x * blockDim.x + threadIdx.x;
    int slot = gid & 31;
    int tile = gid >> 5;
    int ktc  = K >> 5;
    int kt   = tile % ktc;
    int mt   = tile / ktc;
    int m    = mt * 16 + (slot & 15);
    int g    = slot >> 4;
    v16h v;
    const float* row = src + (size_t)m * K;
#pragma unroll
    for (int ch = 0; ch < 2; ++ch) {
        int kbase = kt * 32 + ch * 16 + 8 * g;
#pragma unroll
        for (int j = 0; j < 8; ++j)
            v[ch * 8 + j] = (_Float16)row[kbase + j];
    }
    *(v16h*)(dst + (size_t)tile * 512 + slot * 16) = v;
}

// fp32 row-major [K,N] (scaled) -> f16 B-swizzled
__global__ __launch_bounds__(256) void conv_b_swz(const float* __restrict__ src,
                                                  _Float16* __restrict__ dst,
                                                  int K, int N, float scale) {
    int gid  = blockIdx.x * blockDim.x + threadIdx.x;
    int slot = gid & 31;
    int tile = gid >> 5;
    int ktc  = K >> 5;
    int kt   = tile % ktc;
    int nt   = tile / ktc;
    int n    = nt * 16 + (slot & 15);
    int kbase = kt * 32 + 8 * (slot >> 4);
    v16h v;
#pragma unroll
    for (int j = 0; j < 16; ++j)
        v[j] = (_Float16)(src[(size_t)(kbase + j + (j >= 8 ? 8 : 0)) * N + n] * scale);
    *(v16h*)(dst + (size_t)tile * 512 + slot * 16) = v;
}

// Build polynomial basis of Y = A + c*Kst directly in A-swizzled f16 layout.
// Output logical matrix: [B_SZ, KBIG], column 4*d+p holds Y[m,d]^p.
__global__ __launch_bounds__(256) void basis_swz(const float* __restrict__ A,
                                                 const float* __restrict__ Kst,
                                                 float c, _Float16* __restrict__ dst) {
    int gid  = blockIdx.x * blockDim.x + threadIdx.x;
    int slot = gid & 31;
    int tile = gid >> 5;
    const int ktc = KBIG >> 5;               // 64
    int kt   = tile % ktc;
    int mt   = tile / ktc;
    int m    = mt * 16 + (slot & 15);
    int g    = slot >> 4;
    size_t rowOff = (size_t)m * D_SZ;
    v16h v;
#pragma unroll
    for (int ch = 0; ch < 2; ++ch) {
        int kbase = kt * 32 + ch * 16 + 8 * g;   // multiple of 8 -> d-aligned
#pragma unroll
        for (int dd = 0; dd < 2; ++dd) {
            int d   = (kbase >> 2) + dd;
            float y  = A[rowOff + d] + c * Kst[rowOff + d];
            float y2 = y * y;
            int o = ch * 8 + dd * 4;
            v[o + 0] = (_Float16)1.0f;
            v[o + 1] = (_Float16)y;
            v[o + 2] = (_Float16)y2;
            v[o + 3] = (_Float16)(y2 * y);
        }
    }
    *(v16h*)(dst + (size_t)tile * 512 + slot * 16) = v;
}

// C[M,N] = scale * Asw[M,K] @ Bsw[K,N]; one wave computes a 32x32 C tile.
__global__ __launch_bounds__(256) void gemm_wmma(const _Float16* __restrict__ Asw,
                                                 const _Float16* __restrict__ Bsw,
                                                 float* __restrict__ C,
                                                 int N, int K, float scale) {
    const int lane   = threadIdx.x & 31;
    const int waveId = blockIdx.x * (blockDim.x >> 5) + (threadIdx.x >> 5);
    const int nWaves = N >> 5;              // waves along N
    const int tm = waveId / nWaves;         // 32-row C tile
    const int tn = waveId % nWaves;         // 32-col C tile
    const int ktc = K >> 5;

    const _Float16* a0p = Asw + ((size_t)(2 * tm)     * ktc) * 512 + lane * 16;
    const _Float16* a1p = Asw + ((size_t)(2 * tm + 1) * ktc) * 512 + lane * 16;
    const _Float16* b0p = Bsw + ((size_t)(2 * tn)     * ktc) * 512 + lane * 16;
    const _Float16* b1p = Bsw + ((size_t)(2 * tn + 1) * ktc) * 512 + lane * 16;

    v8f acc00 = {}, acc01 = {}, acc10 = {}, acc11 = {};
    for (int kt = 0; kt < ktc; ++kt) {
        size_t o = (size_t)kt * 512;
        v16h a0 = *(const v16h*)(a0p + o);
        v16h a1 = *(const v16h*)(a1p + o);
        v16h b0 = *(const v16h*)(b0p + o);
        v16h b1 = *(const v16h*)(b1p + o);
        acc00 = __builtin_amdgcn_wmma_f32_16x16x32_f16(false, a0, false, b0, (short)0, acc00, false, false);
        acc01 = __builtin_amdgcn_wmma_f32_16x16x32_f16(false, a0, false, b1, (short)0, acc01, false, false);
        acc10 = __builtin_amdgcn_wmma_f32_16x16x32_f16(false, a1, false, b0, (short)0, acc10, false, false);
        acc11 = __builtin_amdgcn_wmma_f32_16x16x32_f16(false, a1, false, b1, (short)0, acc11, false, false);
    }

    // C/D layout: VGPR j, lane L -> m_local = j + 8*(L>>4), n_local = L&15
    const int nl = lane & 15;
    const int mh = (lane >> 4) << 3;
#pragma unroll
    for (int j = 0; j < 8; ++j) {
        int m0 = tm * 32 + mh + j;
        int m1 = m0 + 16;
        int n0 = tn * 32 + nl;
        int n1 = n0 + 16;
        C[(size_t)m0 * N + n0] = acc00[j] * scale;
        C[(size_t)m0 * N + n1] = acc01[j] * scale;
        C[(size_t)m1 * N + n0] = acc10[j] * scale;
        C[(size_t)m1 * N + n1] = acc11[j] * scale;
    }
}

// Acc = A + w*Kc (init=1) or Acc += w*Kc (init=0)
__global__ __launch_bounds__(256) void rk_accum(float* __restrict__ Acc,
                                                const float* __restrict__ A,
                                                const float* __restrict__ Kc,
                                                float w, int init) {
    int i = blockIdx.x * blockDim.x + threadIdx.x;
    float base = init ? A[i] : Acc[i];
    Acc[i] = base + w * Kc[i];
}

extern "C" void kernel_launch(void* const* d_in, const int* in_sizes, int n_in,
                              void* d_out, int out_size, void* d_ws, size_t ws_size,
                              hipStream_t stream) {
    (void)in_sizes; (void)n_in; (void)out_size; (void)ws_size;
    const float* x = (const float*)d_in[0];
    const float* P = (const float*)d_in[1];
    const float* U = (const float*)d_in[2];   // [D,4,D] row-major == [KBIG, D]
    const float* F = (const float*)d_in[3];
    float* out = (float*)d_out;

    char* ws = (char*)d_ws;
    size_t off = 0;
    auto alloc = [&](size_t bytes) -> void* {
        void* p = ws + off;
        off += (bytes + 255) & ~(size_t)255;
        return p;
    };
    _Float16* xh   = (_Float16*)alloc((size_t)B_SZ * D_IN * 2);  // also reused for A_f
    _Float16* Ph   = (_Float16*)alloc((size_t)D_IN * D_SZ * 2);
    _Float16* Uh   = (_Float16*)alloc((size_t)KBIG * D_SZ * 2);
    _Float16* Fh   = (_Float16*)alloc((size_t)D_SZ * D_OUT * 2);
    float*    Abuf = (float*)alloc((size_t)B_SZ * D_SZ * 4);
    float*    Accb = (float*)alloc((size_t)B_SZ * D_SZ * 4);
    float*    Kc   = (float*)alloc((size_t)B_SZ * D_SZ * 4);
    _Float16* Bh   = (_Float16*)alloc((size_t)B_SZ * KBIG * 2);

    const float uScale = 2048.0f;  // 2^11, exact
    const float fScale = 512.0f;   // 2^9, exact

    // --- convert inputs into fragment-swizzled f16 ---
    conv_a_swz<<<(B_SZ * D_IN / 16) / 256, 256, 0, stream>>>(x, xh, D_IN);
    conv_b_swz<<<(D_IN * D_SZ / 16) / 256, 256, 0, stream>>>(P, Ph, D_IN, D_SZ, 1.0f);
    conv_b_swz<<<(KBIG * D_SZ / 16) / 256, 256, 0, stream>>>(U, Uh, KBIG, D_SZ, uScale);
    conv_b_swz<<<(D_SZ * D_OUT / 16) / 256, 256, 0, stream>>>(F, Fh, D_SZ, D_OUT, fScale);

    auto gemm = [&](const _Float16* A_, const _Float16* B_, float* C_, int N, int K, float s) {
        int waves = (B_SZ / 32) * (N / 32);     // 1024 for all shapes here
        gemm_wmma<<<waves / 8, 256, 0, stream>>>(A_, B_, C_, N, K, s);
    };

    // A0 = x @ P
    gemm(xh, Ph, Abuf, D_SZ, D_IN, 1.0f);

    // --- fixed-step RK4 on dA/dt = basis(A) @ U ---
    const int basisBlocks = (B_SZ / 16) * (KBIG / 32) * 32 / 256;  // 1024
    const int accumBlocks = (B_SZ * D_SZ) / 256;                    // 4096
    const float h = 1.0f / (float)NSTEPS;
    const float invU = 1.0f / uScale;

    float* state = Abuf;
    float* acc   = Accb;
    for (int step = 0; step < NSTEPS; ++step) {
        // k1
        basis_swz<<<basisBlocks, 256, 0, stream>>>(state, state, 0.0f, Bh);
        gemm(Bh, Uh, Kc, D_SZ, KBIG, invU);
        rk_accum<<<accumBlocks, 256, 0, stream>>>(acc, state, Kc, h / 6.0f, 1);
        // k2
        basis_swz<<<basisBlocks, 256, 0, stream>>>(state, Kc, 0.5f * h, Bh);
        gemm(Bh, Uh, Kc, D_SZ, KBIG, invU);
        rk_accum<<<accumBlocks, 256, 0, stream>>>(acc, state, Kc, h / 3.0f, 0);
        // k3
        basis_swz<<<basisBlocks, 256, 0, stream>>>(state, Kc, 0.5f * h, Bh);
        gemm(Bh, Uh, Kc, D_SZ, KBIG, invU);
        rk_accum<<<accumBlocks, 256, 0, stream>>>(acc, state, Kc, h / 3.0f, 0);
        // k4
        basis_swz<<<basisBlocks, 256, 0, stream>>>(state, Kc, h, Bh);
        gemm(Bh, Uh, Kc, D_SZ, KBIG, invU);
        rk_accum<<<accumBlocks, 256, 0, stream>>>(acc, state, Kc, h / 6.0f, 0);
        // A <- Acc (pointer ping-pong, deterministic)
        float* t = state; state = acc; acc = t;
    }

    // out = A_f @ F
    conv_a_swz<<<(B_SZ * D_SZ / 16) / 256, 256, 0, stream>>>(state, xh, D_SZ);
    gemm(xh, Fh, out, D_OUT, D_SZ, 1.0f / fScale);
}
